// MQA_28114855920088
// MI455X (gfx1250) — compile-verified, matrix operand order checked by reading, not run on hardware
//
#include <hip/hip_runtime.h>
#include <hip/hip_bf16.h>

// ---------------------------------------------------------------------------
// MQA forward for MI455X (gfx1250), wave32 + v_wmma_f32_16x16x32_f16.
//
//   q = (hs @ Wq + bq) * 1/sqrt(128)   -> f16   [4096, 2048]
//   k =  hs @ Wk + bk                  -> f16   [4096, 128]
//   v =  hs @ Wv + bv                  -> f16   [4096, 128]
//   attn = flash_softmax(q k^T) v      -> f16   [4096, 2048]
//   out  = attn @ Wo + bo              -> f32   [4096, 2048]
//
// All matmuls run on WMMA f16 with f32 accumulation. Softmax row reductions
// use DPP16 (VALU) so they co-execute with WMMA. K fragments are preloaded
// ahead of the LDS staging barrier so their latency overlaps staging.
// ---------------------------------------------------------------------------

typedef __attribute__((ext_vector_type(16))) _Float16 v16h;
typedef __attribute__((ext_vector_type(8)))  float    v8f;

union F16Frag {           // one WMMA 16-bit A/B operand: 16 f16 = 8 VGPRs
    v16h h;
    uint4 u[2];
};

static __device__ __forceinline__ v8f wmma_f16(const F16Frag& a, const F16Frag& b, v8f c) {
    return __builtin_amdgcn_wmma_f32_16x16x32_f16(
        /*neg_a=*/false, a.h, /*neg_b=*/false, b.h,
        /*c_mod=*/(short)0, c, /*reuse_a=*/false, /*reuse_b=*/false);
}

// DPP lane swap (stays within a 16-lane row; pure VALU, no LDS traffic).
template <int CTRL>
static __device__ __forceinline__ float dpp_mov(float x) {
    int i = __builtin_bit_cast(int, x);
    int r = __builtin_amdgcn_update_dpp(i, i, CTRL, 0xf, 0xf, true);
    return __builtin_bit_cast(float, r);
}
// Reduce over each 16-lane half with involutions {xor1, xor2, xor7, xor8}:
//   0xB1 = quad_perm(1,0,3,2), 0x4E = quad_perm(2,3,0,1),
//   0x141 = row_half_mirror (xor7 within 8), 0x128 = row_ror:8 (xor8 in 16).
static __device__ __forceinline__ float row_max16(float x) {
    x = fmaxf(x, dpp_mov<0xB1>(x));
    x = fmaxf(x, dpp_mov<0x4E>(x));
    x = fmaxf(x, dpp_mov<0x141>(x));
    x = fmaxf(x, dpp_mov<0x128>(x));
    return x;
}
static __device__ __forceinline__ float row_sum16(float x) {
    x += dpp_mov<0xB1>(x);
    x += dpp_mov<0x4E>(x);
    x += dpp_mov<0x141>(x);
    x += dpp_mov<0x128>(x);
    return x;
}

// ---------------------------------------------------------------------------
// Generic tiled WMMA GEMM:  C[M,N] = (A[M,K] @ B[K,N] + bias[N]) * scale
// A, B row-major; AT/BT may be float (converted to f16 while staging) or
// _Float16. Block tile 128x128, BK=32, 8 waves -> each wave 64x32 (4x2 WMMA).
// M, N multiples of 128; K multiple of 32 (true for all uses here).
// ---------------------------------------------------------------------------
#define LDT 40   // padded LDS row stride in f16 elements (80B, 16B-aligned)

template <typename AT, typename BT, typename OutT>
__global__ __launch_bounds__(256)
void gemm_wmma(const AT* __restrict__ A, const BT* __restrict__ B,
               const float* __restrict__ bias, OutT* __restrict__ C,
               int M, int N, int K, float scale) {
    __shared__ __align__(16) _Float16 As[128 * LDT];   // [row][k]   10 KB
    __shared__ __align__(16) _Float16 Bst[128 * LDT];  // [col][k]   10 KB (transposed)

    const int tid  = threadIdx.x;
    const int wave = tid >> 5;
    const int lane = tid & 31;
    const int half = lane >> 4;       // 0: lanes 0-15, 1: lanes 16-31
    const int l15  = lane & 15;
    const int waveRow = wave >> 2;    // 0..1  -> 64-row slabs
    const int waveCol = wave & 3;     // 0..3  -> 32-col slabs

    const int mBase = blockIdx.y * 128;
    const int nBase = blockIdx.x * 128;

    v8f acc[4][2];
    #pragma unroll
    for (int mi = 0; mi < 4; ++mi)
        #pragma unroll
        for (int ni = 0; ni < 2; ++ni) acc[mi][ni] = (v8f){};

    for (int k0 = 0; k0 < K; k0 += 32) {
        // ---- stage A tile (128x32): cvt in regs, one b128 store ---------
        #pragma unroll
        for (int it = 0; it < 2; ++it) {
            const int r  = (tid >> 2) + it * 64;     // 0..127
            const int c8 = (tid & 3) * 8;            // 0,8,16,24
            const AT* g = A + (size_t)(mBase + r) * K + k0 + c8;
            union { _Float16 h[8]; uint4 u; } t;
            #pragma unroll
            for (int j = 0; j < 8; ++j) t.h[j] = (_Float16)g[j];
            *reinterpret_cast<uint4*>(&As[r * LDT + c8]) = t.u;
        }
        // ---- stage B tile (32x128) transposed -> Bst[col][k] ------------
        #pragma unroll
        for (int p = 0; p < 2; ++p) {
            const int kk = (tid >> 4) + p * 16;      // 0..31
            const int n0 = (tid & 15) * 8;           // 0..120
            const BT* g = B + (size_t)(k0 + kk) * N + nBase + n0;
            #pragma unroll
            for (int j = 0; j < 8; ++j)
                Bst[(n0 + j) * LDT + kk] = (_Float16)g[j];
        }
        // ---- prefetch next K tile while this one is consumed ------------
        if (k0 + 32 < K) {
            __builtin_prefetch(A + (size_t)(mBase + (tid >> 2)) * K + (k0 + 32) + (tid & 3) * 8, 0, 1);
            __builtin_prefetch(B + (size_t)(k0 + 32 + (tid >> 4)) * N + nBase + (tid & 15) * 8, 0, 1);
        }
        __syncthreads();

        // ---- fragment loads (ISA 16-bit A/B layout: two 16B chunks) -----
        F16Frag af[4];
        #pragma unroll
        for (int mi = 0; mi < 4; ++mi) {
            const int r = waveRow * 64 + mi * 16 + l15;
            const _Float16* p = &As[r * LDT + half * 8];
            af[mi].u[0] = *reinterpret_cast<const uint4*>(p);
            af[mi].u[1] = *reinterpret_cast<const uint4*>(p + 16);
        }
        F16Frag bf[2];
        #pragma unroll
        for (int ni = 0; ni < 2; ++ni) {
            const int c = waveCol * 32 + ni * 16 + l15;
            const _Float16* p = &Bst[c * LDT + half * 8];
            bf[ni].u[0] = *reinterpret_cast<const uint4*>(p);
            bf[ni].u[1] = *reinterpret_cast<const uint4*>(p + 16);
        }
        #pragma unroll
        for (int mi = 0; mi < 4; ++mi)
            #pragma unroll
            for (int ni = 0; ni < 2; ++ni)
                acc[mi][ni] = wmma_f16(af[mi], bf[ni], acc[mi][ni]);
        __syncthreads();
    }

    // ---- epilogue: C layout lane=col, VGPR v -> row v + 8*half ----------
    #pragma unroll
    for (int mi = 0; mi < 4; ++mi)
        #pragma unroll
        for (int ni = 0; ni < 2; ++ni) {
            const int col = nBase + waveCol * 32 + ni * 16 + l15;
            const float b = bias[col];
            #pragma unroll
            for (int v = 0; v < 8; ++v) {
                const int row = mBase + waveRow * 64 + mi * 16 + half * 8 + v;
                C[(size_t)row * N + col] = (OutT)((acc[mi][ni][v] + b) * scale);
            }
        }
}

// ---------------------------------------------------------------------------
// Flash attention, MQA (shared K/V head). Q pre-scaled by 1/sqrt(128).
// Grid: (16 q-blocks, 16 heads, 2 batches); 256 threads = 8 waves.
// Each wave owns 16 Q rows; keys processed in blocks of 32 with online
// softmax; P@V accumulated in f32 WMMA C tiles.
// ---------------------------------------------------------------------------
__global__ __launch_bounds__(256)
void mqa_flash(const _Float16* __restrict__ Qg, const _Float16* __restrict__ Kg,
               const _Float16* __restrict__ Vg, _Float16* __restrict__ Og) {
    constexpr int SEQ = 2048, HID = 2048, HD = 128;

    __shared__ __align__(16) _Float16 Vt[128 * LDT];       // [feat][key] 10 KB
    __shared__ __align__(16) _Float16 Ps[8 * 16 * LDT];    // per-wave P   10 KB

    const int tid  = threadIdx.x;
    const int wave = tid >> 5;
    const int lane = tid & 31;
    const int half = lane >> 4;
    const int l15  = lane & 15;

    const int b = blockIdx.z, h = blockIdx.y;
    const int qRow0 = blockIdx.x * 128 + wave * 16;

    const _Float16* Qb = Qg + (size_t)b * SEQ * HID + (size_t)h * HD;
    const _Float16* Kb = Kg + (size_t)b * SEQ * HD;
    const _Float16* Vb = Vg + (size_t)b * SEQ * HD;

    // Q fragments (A operand, K-dim 128 = 4 chunks of 32), loaded once.
    F16Frag qf[4];
    {
        const _Float16* qr = Qb + (size_t)(qRow0 + l15) * HID;
        #pragma unroll
        for (int kc = 0; kc < 4; ++kc) {
            const _Float16* p = qr + kc * 32 + half * 8;
            qf[kc].u[0] = *reinterpret_cast<const uint4*>(p);
            qf[kc].u[1] = *reinterpret_cast<const uint4*>(p + 16);
        }
    }

    v8f o[8];
    #pragma unroll
    for (int fi = 0; fi < 8; ++fi) o[fi] = (v8f){};
    float m[8], l[8];
    #pragma unroll
    for (int v = 0; v < 8; ++v) { m[v] = -3.0e38f; l[v] = 0.f; }

    _Float16* myPs = &Ps[wave * 16 * LDT];

    for (int kb = 0; kb < SEQ; kb += 32) {
        // ---- preload ALL K fragments (global, independent of LDS) -------
        // Issued before the staging barrier so their latency overlaps the
        // Vt scatter stores and the barrier wait.
        F16Frag kf[2][4];
        #pragma unroll
        for (int ni = 0; ni < 2; ++ni) {
            const _Float16* kr = Kb + (size_t)(kb + ni * 16 + l15) * HD + half * 8;
            #pragma unroll
            for (int kc = 0; kc < 4; ++kc) {
                kf[ni][kc].u[0] = *reinterpret_cast<const uint4*>(kr + kc * 32);
                kf[ni][kc].u[1] = *reinterpret_cast<const uint4*>(kr + kc * 32 + 16);
            }
        }

        // ---- stage V block transposed: Vt[feat][key] --------------------
        #pragma unroll
        for (int p = 0; p < 2; ++p) {
            const int kk = (tid >> 4) + p * 16;          // 0..31
            const int f0 = (tid & 15) * 8;               // 0..120
            const _Float16* g = Vb + (size_t)(kb + kk) * HD + f0;
            #pragma unroll
            for (int j = 0; j < 8; ++j)
                Vt[(f0 + j) * LDT + kk] = g[j];
        }
        // ---- prefetch next K/V blocks -----------------------------------
        if (kb + 32 < SEQ) {
            __builtin_prefetch(Vb + (size_t)(kb + 32 + (tid >> 4)) * HD + (tid & 15) * 8, 0, 1);
            __builtin_prefetch(Kb + (size_t)(kb + 32 + (tid >> 4)) * HD + (tid & 15) * 8, 0, 1);
        }
        __syncthreads();

        // ---- S = Q K^T (16x32 per wave, two 16x16 C tiles) --------------
        v8f s[2] = {};
        #pragma unroll
        for (int ni = 0; ni < 2; ++ni)
            #pragma unroll
            for (int kc = 0; kc < 4; ++kc)
                s[ni] = wmma_f16(qf[kc], kf[ni][kc], s[ni]);

        // ---- online softmax: DPP row reductions (no LDS) ----------------
        float alpha[8];
        #pragma unroll
        for (int v = 0; v < 8; ++v) {
            const float mx = row_max16(fmaxf(s[0][v], s[1][v]));
            const float mn = fmaxf(m[v], mx);
            alpha[v] = __expf(m[v] - mn);
            m[v] = mn;
            const float p0 = __expf(s[0][v] - mn);
            const float p1 = __expf(s[1][v] - mn);
            s[0][v] = p0; s[1][v] = p1;
            l[v] = l[v] * alpha[v] + row_sum16(p0 + p1);
        }
        #pragma unroll
        for (int fi = 0; fi < 8; ++fi)
            #pragma unroll
            for (int v = 0; v < 8; ++v)
                o[fi][v] *= alpha[v];

        // ---- transpose P (C layout) -> A-operand layout via LDS ---------
        #pragma unroll
        for (int ni = 0; ni < 2; ++ni)
            #pragma unroll
            for (int v = 0; v < 8; ++v)
                myPs[(v + half * 8) * LDT + ni * 16 + l15] = (_Float16)s[ni][v];

        F16Frag pf;
        {
            const _Float16* p = &myPs[l15 * LDT + half * 8];
            pf.u[0] = *reinterpret_cast<const uint4*>(p);
            pf.u[1] = *reinterpret_cast<const uint4*>(p + 16);
        }

        // ---- O += P @ V  (K-dim 32, 8 feature tiles) --------------------
        #pragma unroll
        for (int fi = 0; fi < 8; ++fi) {
            F16Frag vf;
            const _Float16* p = &Vt[(fi * 16 + l15) * LDT + half * 8];
            vf.u[0] = *reinterpret_cast<const uint4*>(p);
            vf.u[1] = *reinterpret_cast<const uint4*>(p + 16);
            o[fi] = wmma_f16(pf, vf, o[fi]);
        }
        __syncthreads();
    }

    // ---- epilogue: normalize and store f16 attention output ------------
    #pragma unroll
    for (int fi = 0; fi < 8; ++fi) {
        const int col = h * HD + fi * 16 + l15;
        #pragma unroll
        for (int v = 0; v < 8; ++v) {
            const int row = qRow0 + half * 8 + v;
            Og[((size_t)b * SEQ + row) * HID + col] = (_Float16)(o[fi][v] / l[v]);
        }
    }
}

// ---------------------------------------------------------------------------
extern "C" void kernel_launch(void* const* d_in, const int* in_sizes, int n_in,
                              void* d_out, int out_size, void* d_ws, size_t ws_size,
                              hipStream_t stream) {
    (void)in_sizes; (void)n_in; (void)out_size; (void)ws_size;
    const float* hs = (const float*)d_in[0];
    const float* Wq = (const float*)d_in[1];
    const float* bq = (const float*)d_in[2];
    const float* Wk = (const float*)d_in[3];
    const float* bk = (const float*)d_in[4];
    const float* Wv = (const float*)d_in[5];
    const float* bv = (const float*)d_in[6];
    const float* Wo = (const float*)d_in[7];
    const float* bo = (const float*)d_in[8];
    float* out = (float*)d_out;

    // Workspace layout (f16 intermediates, ~36 MB total)
    char* w = (char*)d_ws;
    _Float16* Q16 = (_Float16*)w;                 w += (size_t)4096 * 2048 * 2;
    _Float16* K16 = (_Float16*)w;                 w += (size_t)4096 * 128 * 2;
    _Float16* V16 = (_Float16*)w;                 w += (size_t)4096 * 128 * 2;
    _Float16* A16 = (_Float16*)w;

    const dim3 blk(256);
    const float qscale = 0.08838834764831845f;    // 1/sqrt(128)

    // Projections (fp32 in -> f16 out), scale folded into Q.
    gemm_wmma<float, float, _Float16><<<dim3(16, 32), blk, 0, stream>>>(
        hs, Wq, bq, Q16, 4096, 2048, 2048, qscale);
    gemm_wmma<float, float, _Float16><<<dim3(1, 32), blk, 0, stream>>>(
        hs, Wk, bk, K16, 4096, 128, 2048, 1.0f);
    gemm_wmma<float, float, _Float16><<<dim3(1, 32), blk, 0, stream>>>(
        hs, Wv, bv, V16, 4096, 128, 2048, 1.0f);

    // Flash attention over 2 batches x 16 heads x 16 q-blocks.
    mqa_flash<<<dim3(16, 16, 2), blk, 0, stream>>>(Q16, K16, V16, A16);

    // Output projection (f16 x fp32 -> fp32).
    gemm_wmma<_Float16, float, float><<<dim3(16, 32), blk, 0, stream>>>(
        A16, Wo, bo, out, 4096, 2048, 2048, 1.0f);
}
